// TFLSTM_24515673326155
// MI455X (gfx1250) — matrix operand order repaired and checked by reading, not compiled, not run on hardware
//
#include <hip/hip_runtime.h>
#include <hip/hip_bf16.h>
#include <cstdint>
#include <cstddef>

// ---------------------------------------------------------------------------
// Persistent LSTMP kernel for MI455X (gfx1250, wave32, WMMA).
//
// 128 WGs, one per WGP. WG g owns hidden units [16g,16g+16) -> 64 gate rows
// (4 gates x 16); their f16 weights (64x1024 = 128KB) stay resident in LDS
// for the whole 256-step sequence. Gate GEMM per step: M=64 x N=64 x K=1024
// with v_wmma_f32_16x16x32_f16; A=[x_t|h_{t-1}] streamed in 64-wide K chunks,
// double-buffered via CDNA5 async global->LDS copies (ASYNCcnt) so the DMA of
// chunk kc+1 overlaps the WMMAs of chunk kc. Peephole cell update is WG-local
// (c resident in LDS). Projection (WGs 0..31, 16 P-columns each, Wp slice
// resident in LDS): K=2048 split across the 8 waves with an LDS reduction,
// A-fragments from global software-pipelined at depth 4. Steps separated by a
// grid-wide sense-reversing barrier (device atomics + s_sleep); assumes the
// 128 WGs are co-resident (~233KB LDS each -> 1 WG/WGP).
// ---------------------------------------------------------------------------

typedef __attribute__((ext_vector_type(16))) _Float16 v16h;
typedef __attribute__((ext_vector_type(8)))  _Float16 v8h;
typedef __attribute__((ext_vector_type(8)))  float    v8f;

#if __has_builtin(__builtin_amdgcn_global_load_async_to_lds_b128)
#define USE_ASYNC_LDS 1
#else
#define USE_ASYNC_LDS 0
#endif

namespace {
constexpr int S_LEN = 256, BATCH = 64, DIN = 512, HID = 2048, PROJ = 512, NLAYER = 2;
constexpr int NWG = 128, NTHREADS = 256, NPROJ_WG = 32;
constexpr int GROWS = 64;              // gate rows per WG
constexpr int KTOT = DIN + PROJ;       // 1024
constexpr int KCHUNK = 64, NCHUNKS = KTOT / KCHUNK;
// LDS row strides (elements), padded to dodge bank conflicts
constexpr int WT_LD = 1032;            // gate W tile rows (f16)
constexpr int WP_LD = 2056;            // Wp slice rows (f16)
constexpr int AC_LD = 72;              // A chunk rows (f16)
constexpr int GB_LD = 68;              // gate buffer rows (f32)
constexpr int CB_LD = 68;              // c buffer rows (f32)
constexpr size_t AC_ELEMS = (size_t)BATCH * AC_LD;    // one A-chunk buffer (elems)
constexpr size_t SM_WT = (size_t)GROWS * WT_LD * 2;
constexpr size_t SM_WP = (size_t)16 * WP_LD * 2;
constexpr size_t SM_AC = AC_ELEMS * 2;                // bytes of one buffer
constexpr size_t SM_GB = (size_t)GROWS * GB_LD * 4;
constexpr size_t SM_CB = (size_t)16 * CB_LD * 4;
constexpr size_t SMEM_BYTES = SM_WT + SM_WP + 2 * SM_AC + SM_GB + SM_CB; // ~233KB
}

// ---- WMMA fragment helpers (CDNA5 16x16x32 f16 layouts, ISA 7.12.2) -------
__device__ __forceinline__ v16h load_frag_A(const _Float16* base, int ldh, int lane) {
    const _Float16* p = base + (size_t)(lane & 15) * ldh + ((lane < 16) ? 0 : 8);
    v8h lo = *(const v8h*)(p);
    v8h hi = *(const v8h*)(p + 16);
    v16h r;
#pragma unroll
    for (int i = 0; i < 8; ++i) { r[i] = lo[i]; r[i + 8] = hi[i]; }
    return r;
}
__device__ __forceinline__ v16h load_frag_B(const _Float16* base, int ldh, int lane) {
    const _Float16* p = base + (size_t)(lane & 15) * ldh + ((lane < 16) ? 0 : 8);
    v8h lo = *(const v8h*)(p);
    v8h hi = *(const v8h*)(p + 16);
    v16h r;
#pragma unroll
    for (int i = 0; i < 8; ++i) { r[i] = lo[i]; r[i + 8] = hi[i]; }
    return r;
}
__device__ __forceinline__ v8f wmma16(v16h a, v16h b, v8f c) {
    return __builtin_amdgcn_wmma_f32_16x16x32_f16(false, a, false, b, (short)0, c,
                                                  false, false);
}
__device__ __forceinline__ float sigm(float x) { return 1.0f / (1.0f + expf(-x)); }

// ---- CDNA5 async global->LDS copy (ASYNCcnt) -------------------------------
#if USE_ASYNC_LDS
// Builtin signature (from hipcc diagnostic): param0 = v4i* in AS1 ("__device__"),
// param1 = v4i* in AS3 (LDS), then imm offset, imm cpol.
typedef int v4i __attribute__((__vector_size__(4 * sizeof(int))));
typedef __attribute__((address_space(1))) v4i as1_v4i;
typedef __attribute__((address_space(3))) v4i as3_v4i;
template <int OFF>
__device__ __forceinline__ void async_copy_b128(const _Float16* g, _Float16* l) {
    // generic LDS pointer low 32 bits == LDS offset; global generic == full VA
    __builtin_amdgcn_global_load_async_to_lds_b128(
        (as1_v4i*)(uintptr_t)g, (as3_v4i*)(uint32_t)(uintptr_t)l, OFF, 0);
}
__device__ __forceinline__ void wait_async0() {
#if __has_builtin(__builtin_amdgcn_s_wait_asynccnt)
    __builtin_amdgcn_s_wait_asynccnt(0);
#else
    asm volatile("s_wait_asynccnt 0x0" ::: "memory");
#endif
}
#endif

// ---- grid-wide sense/generation barrier ------------------------------------
__device__ __forceinline__ void grid_barrier(unsigned* cnt, unsigned* gen) {
    __syncthreads();
    if (threadIdx.x == 0) {
        __threadfence();
        unsigned g = __atomic_load_n(gen, __ATOMIC_RELAXED);
        unsigned arrived = atomicAdd(cnt, 1u) + 1u;
        if (arrived == (unsigned)gridDim.x) {
            __atomic_store_n(cnt, 0u, __ATOMIC_RELAXED);
            __atomic_store_n(gen, g + 1u, __ATOMIC_RELEASE);
        } else {
            while (__atomic_load_n(gen, __ATOMIC_ACQUIRE) == g)
                __builtin_amdgcn_s_sleep(2);
        }
        __threadfence();
    }
    __syncthreads();
}

// ---- prologue kernels ------------------------------------------------------
__global__ void cvt_f32_to_f16(const float* __restrict__ in, _Float16* __restrict__ out,
                               int n) {
    int i = blockIdx.x * blockDim.x + threadIdx.x;
    if (i < n) out[i] = (_Float16)in[i];
}
// Permute W rows so WG g's 64 rows {gate*16+u} are contiguous.
__global__ void permute_W_f16(const float* __restrict__ W, _Float16* __restrict__ Wg) {
    int bid = blockIdx.x;           // [0, 2*8192)
    int l = bid >> 13;
    int r = bid & 8191;
    int gate = r >> 11;
    int hid = r & 2047;
    int g = hid >> 4, u = hid & 15;
    size_t src = ((size_t)l * 8192 + r) * KTOT;
    size_t dst = ((size_t)l * 8192 + (size_t)g * 64 + gate * 16 + u) * KTOT;
    for (int k = threadIdx.x; k < KTOT; k += blockDim.x)
        Wg[dst + k] = (_Float16)W[src + k];
}

// ---- persistent recurrence kernel -----------------------------------------
__global__ void lstmp_persistent(
    const _Float16* __restrict__ x0,    // [S][B][DIN] f16
    _Float16* __restrict__ x1,          // [S][B][PROJ] f16 (layer0 out / layer1 in)
    const _Float16* __restrict__ Wg,    // [L][8192][1024] f16, row-permuted
    const _Float16* __restrict__ Wpw,   // [L][512][2048] f16
    const float* __restrict__ bias,     // [L][8192]
    const float* __restrict__ wi, const float* __restrict__ wf,
    const float* __restrict__ wo,       // [L][H] peepholes
    _Float16* __restrict__ h16,         // [B][PROJ]
    _Float16* __restrict__ m16,         // [B][HID]
    float* __restrict__ out,            // [B][S][P]
    float* __restrict__ hfin,           // [L][B][P]
    float* __restrict__ cfin,           // [L][B][H]
    unsigned* __restrict__ ctrl) {
    extern __shared__ __align__(16) char smem_raw[];
    _Float16* wtile   = (_Float16*)smem_raw;                          // [64][WT_LD]
    _Float16* wptile  = (_Float16*)(smem_raw + SM_WT);                // [16][WP_LD]
    _Float16* achunk0 = (_Float16*)(smem_raw + SM_WT + SM_WP);        // [2][64][AC_LD]
    float*    gbuf    = (float*)(smem_raw + SM_WT + SM_WP + 2 * SM_AC);
    float*    cbuf    = (float*)(smem_raw + SM_WT + SM_WP + 2 * SM_AC + SM_GB);

    const int wg = blockIdx.x;
    const int tid = threadIdx.x;
    const int lane = tid & 31;
    const int wave = tid >> 5;
    unsigned* bar_cnt = ctrl;
    unsigned* bar_gen = ctrl + 16;

    const int st_row = tid >> 2;            // staging: 64 rows x 4 segments
    const int st_seg = (tid & 3) * 16;
    // buffer selection by offset arithmetic from ONE LDS base, so address-space
    // inference keeps all A-fragment reads on the ds_load path (no flat_load).
    _Float16* const st_dst0 = achunk0 + st_row * AC_LD + st_seg;

    for (int l = 0; l < NLAYER; ++l) {
        // ---- per-layer init: stage resident LDS tiles, zero state ----
        {
            const _Float16* src = Wg + ((size_t)l * 8192 + (size_t)wg * GROWS) * KTOT;
            for (int i = tid; i < GROWS * KTOT / 8; i += NTHREADS) {
                int row = i >> 7;
                int c8  = (i & 127) * 8;
                *(v8h*)(wtile + (size_t)row * WT_LD + c8) =
                    *(const v8h*)(src + (size_t)row * KTOT + c8);
            }
        }
        if (wg < NPROJ_WG) {
            const _Float16* src = Wpw + ((size_t)l * PROJ + (size_t)wg * 16) * HID;
            for (int i = tid; i < 16 * HID / 8; i += NTHREADS) {
                int row = i >> 8;
                int c8  = (i & 255) * 8;
                *(v8h*)(wptile + (size_t)row * WP_LD + c8) =
                    *(const v8h*)(src + (size_t)row * HID + c8);
            }
            for (int i = tid; i < 2 * PROJ; i += NTHREADS) {     // h_0 = 0
                int b = 2 * wg + (i >= PROJ ? 1 : 0);
                h16[(size_t)b * PROJ + (i & (PROJ - 1))] = (_Float16)0.0f;
            }
        }
        for (int i = tid; i < 16 * BATCH; i += NTHREADS)         // c_0 = 0
            cbuf[(i >> 6) * CB_LD + (i & 63)] = 0.0f;
        grid_barrier(bar_cnt, bar_gen);

        const _Float16* xin = l ? x1 : x0;
        const float* bl  = bias + (size_t)l * 4 * HID;
        const float* wil = wi + (size_t)l * HID;
        const float* wfl = wf + (size_t)l * HID;
        const float* wol = wo + (size_t)l * HID;

        for (int t = 0; t < S_LEN; ++t) {
            // ============== phase 1: gate GEMM + cell update ==============
            const int mt  = wave & 3;
            const int nt0 = (wave >> 2) * 2;
            v8f acc0 = {}; v8f acc1 = {};

            // chunk source: cols [kc*64, kc*64+64) of A=[x_t | h_{t-1}]
            auto csrc = [&](int kc) -> const _Float16* {
                return (kc < 8)
                    ? xin + ((size_t)t * BATCH + st_row) * DIN + kc * KCHUNK + st_seg
                    : h16 + (size_t)st_row * PROJ + (kc - 8) * KCHUNK + st_seg;
            };
            // prime chunk 0 into buffer 0
            {
#if USE_ASYNC_LDS
                async_copy_b128<0>(csrc(0), st_dst0);
                async_copy_b128<16>(csrc(0), st_dst0);
                wait_async0();
#else
                const _Float16* s = csrc(0);
                *(v8h*)st_dst0 = *(const v8h*)s;
                *(v8h*)(st_dst0 + 8) = *(const v8h*)(s + 8);
#endif
            }
            __syncthreads();

            for (int kc = 0; kc < NCHUNKS; ++kc) {
                const size_t curoff = (size_t)(kc & 1) * AC_ELEMS;
                const size_t nxtoff = (size_t)((kc + 1) & 1) * AC_ELEMS;
                const bool more = (kc + 1 < NCHUNKS);
                // issue staging of chunk kc+1 before computing chunk kc
#if USE_ASYNC_LDS
                if (more) {
                    const _Float16* s = csrc(kc + 1);
                    async_copy_b128<0>(s, st_dst0 + nxtoff);
                    async_copy_b128<16>(s, st_dst0 + nxtoff);
                }
#else
                v8h s0 = {}, s1 = {};
                if (more) {                         // loads issued; waited at store
                    const _Float16* s = csrc(kc + 1);
                    s0 = *(const v8h*)s;
                    s1 = *(const v8h*)(s + 8);
                }
#endif
#pragma unroll
                for (int kk = 0; kk < KCHUNK; kk += 32) {
                    v16h a  = load_frag_A(achunk0 + curoff + (mt * 16) * AC_LD + kk,
                                          AC_LD, lane);
                    v16h b0 = load_frag_B(wtile + (size_t)(nt0 * 16) * WT_LD +
                                          kc * KCHUNK + kk, WT_LD, lane);
                    v16h b1 = load_frag_B(wtile + (size_t)((nt0 + 1) * 16) * WT_LD +
                                          kc * KCHUNK + kk, WT_LD, lane);
                    acc0 = wmma16(a, b0, acc0);
                    acc1 = wmma16(a, b1, acc1);
                }
#if USE_ASYNC_LDS
                if (more) wait_async0();
#else
                if (more) {
                    *(v8h*)(st_dst0 + nxtoff) = s0;
                    *(v8h*)(st_dst0 + nxtoff + 8) = s1;
                }
#endif
                __syncthreads();
            }
            // scatter D tiles to gbuf[gate_row][batch]
            {
                int n = lane & 15;
                int mb = mt * 16 + ((lane < 16) ? 0 : 8);
#pragma unroll
                for (int r = 0; r < 8; ++r) {
                    gbuf[(nt0 * 16 + n) * GB_LD + mb + r]       = acc0[r];
                    gbuf[((nt0 + 1) * 16 + n) * GB_LD + mb + r] = acc1[r];
                }
            }
            __syncthreads();
            // peephole LSTM cell update for this WG's 16 hidden units
            for (int i = tid; i < 16 * BATCH; i += NTHREADS) {
                int u = i >> 6, b = i & 63;
                int hg = wg * 16 + u;
                float ig = gbuf[(0 * 16 + u) * GB_LD + b] + bl[0 * HID + hg];
                float jg = gbuf[(1 * 16 + u) * GB_LD + b] + bl[1 * HID + hg];
                float fg = gbuf[(2 * 16 + u) * GB_LD + b] + bl[2 * HID + hg];
                float og = gbuf[(3 * 16 + u) * GB_LD + b] + bl[3 * HID + hg];
                float c_old = cbuf[u * CB_LD + b];
                float ft = sigm(fg + wfl[hg] * c_old + 1.0f);   // FORGET_BIAS
                float it = sigm(ig + wil[hg] * c_old);
                float cn = ft * c_old + it * tanhf(jg);
                float ot = sigm(og + wol[hg] * cn);
                float mv = ot * tanhf(cn);
                cbuf[u * CB_LD + b] = cn;
                m16[(size_t)b * HID + hg] = (_Float16)mv;
                if (t == S_LEN - 1)
                    cfin[((size_t)l * BATCH + b) * HID + hg] = cn;
            }
            grid_barrier(bar_cnt, bar_gen);

            // ============== phase 2: projection h = m @ Wp^T ==============
            // All 8 waves: mt2 = wave&3 (batch tile), kh = wave>>2 (K half).
            // Depth-4 software pipeline on A-fragment global loads.
            if (wg < NPROJ_WG) {
                const int mt2 = wave & 3;
                const int kh  = wave >> 2;
                const _Float16* abase =
                    m16 + (size_t)(mt2 * 16) * HID + kh * (HID / 2);
                const _Float16* bbase = wptile + kh * (HID / 2);
                v8f acc = {};
                v16h a0 = load_frag_A(abase + 0,  HID, lane);
                v16h a1 = load_frag_A(abase + 32, HID, lane);
                v16h a2 = load_frag_A(abase + 64, HID, lane);
                v16h a3 = load_frag_A(abase + 96, HID, lane);
#pragma unroll
                for (int k0 = 0; k0 < HID / 2 - 128; k0 += 128) {
                    v16h n0 = load_frag_A(abase + k0 + 128, HID, lane);
                    v16h n1 = load_frag_A(abase + k0 + 160, HID, lane);
                    v16h n2 = load_frag_A(abase + k0 + 192, HID, lane);
                    v16h n3 = load_frag_A(abase + k0 + 224, HID, lane);
                    acc = wmma16(a0, load_frag_B(bbase + k0,      WP_LD, lane), acc);
                    acc = wmma16(a1, load_frag_B(bbase + k0 + 32, WP_LD, lane), acc);
                    acc = wmma16(a2, load_frag_B(bbase + k0 + 64, WP_LD, lane), acc);
                    acc = wmma16(a3, load_frag_B(bbase + k0 + 96, WP_LD, lane), acc);
                    a0 = n0; a1 = n1; a2 = n2; a3 = n3;
                }
                {   // last 128-wide block
                    const int k0 = HID / 2 - 128;
                    acc = wmma16(a0, load_frag_B(bbase + k0,      WP_LD, lane), acc);
                    acc = wmma16(a1, load_frag_B(bbase + k0 + 32, WP_LD, lane), acc);
                    acc = wmma16(a2, load_frag_B(bbase + k0 + 64, WP_LD, lane), acc);
                    acc = wmma16(a3, load_frag_B(bbase + k0 + 96, WP_LD, lane), acc);
                }
                // reduce the two K-halves through LDS (reuse gbuf)
                int n = lane & 15;
                int mloc = (lane < 16) ? 0 : 8;
                if (kh == 1) {
                    float* dst = gbuf + mt2 * 256;
#pragma unroll
                    for (int r = 0; r < 8; ++r) dst[(mloc + r) * 16 + n] = acc[r];
                }
                __syncthreads();
                if (kh == 0) {
                    const float* srcr = gbuf + mt2 * 256;
                    int p = wg * 16 + n;
#pragma unroll
                    for (int r = 0; r < 8; ++r) {
                        int b = mt2 * 16 + mloc + r;
                        float v = acc[r] + srcr[(mloc + r) * 16 + n];
                        h16[(size_t)b * PROJ + p] = (_Float16)v;
                        if (l == 0)
                            x1[((size_t)t * BATCH + b) * PROJ + p] = (_Float16)v;
                        else
                            out[((size_t)b * S_LEN + t) * PROJ + p] = v;
                        if (t == S_LEN - 1)
                            hfin[((size_t)l * BATCH + b) * PROJ + p] = v;
                    }
                }
            }
            grid_barrier(bar_cnt, bar_gen);
        } // t
    } // layer
}

// ---------------------------------------------------------------------------
extern "C" void kernel_launch(void* const* d_in, const int* in_sizes, int n_in,
                              void* d_out, int out_size, void* d_ws, size_t ws_size,
                              hipStream_t stream) {
    const float* x  = (const float*)d_in[0];  // [256][64][512]
    const float* W  = (const float*)d_in[1];  // [2][8192][1024]
    const float* b  = (const float*)d_in[2];  // [2][8192]
    const float* wi = (const float*)d_in[3];  // [2][2048]
    const float* wf = (const float*)d_in[4];
    const float* wo = (const float*)d_in[5];
    const float* Wp = (const float*)d_in[6];  // [2][512][2048]

    float* out  = (float*)d_out;                          // [64][256][512]
    float* hfin = out + (size_t)BATCH * S_LEN * PROJ;     // [2][64][512]
    float* cfin = hfin + (size_t)NLAYER * BATCH * PROJ;   // [2][64][2048]

    char* ws = (char*)d_ws;
    size_t off = 0;
    auto carve = [&](size_t bytes) {
        void* p = ws + off;
        off += (bytes + 255) & ~(size_t)255;
        return p;
    };
    unsigned*  ctrl = (unsigned*)carve(256);
    _Float16*  x0f  = (_Float16*)carve((size_t)S_LEN * BATCH * DIN * 2);
    _Float16*  x1f  = (_Float16*)carve((size_t)S_LEN * BATCH * PROJ * 2);
    _Float16*  Wgf  = (_Float16*)carve((size_t)NLAYER * 4 * HID * KTOT * 2);
    _Float16*  Wpf  = (_Float16*)carve((size_t)NLAYER * PROJ * HID * 2);
    _Float16*  h16  = (_Float16*)carve((size_t)BATCH * PROJ * 2);
    _Float16*  m16  = (_Float16*)carve((size_t)BATCH * HID * 2);

    (void)hipMemsetAsync(ctrl, 0, 256, stream);   // barrier count + generation

    {
        int n = S_LEN * BATCH * DIN;
        cvt_f32_to_f16<<<n / 256, 256, 0, stream>>>(x, x0f, n);
    }
    permute_W_f16<<<NLAYER * 4 * HID, 256, 0, stream>>>(W, Wgf);
    {
        int n = NLAYER * PROJ * HID;
        cvt_f32_to_f16<<<n / 256, 256, 0, stream>>>(Wp, Wpf, n);
    }

    lstmp_persistent<<<NWG, NTHREADS, SMEM_BYTES, stream>>>(
        x0f, x1f, Wgf, Wpf, b, wi, wf, wo, h16, m16, out, hfin, cfin, ctrl);
}
